// SE3DiffusionTransformer_45827301048670
// MI455X (gfx1250) — matrix operand order repaired and checked by reading, not compiled, not run on hardware
//
#include <hip/hip_runtime.h>
#include <hip/hip_bf16.h>

#define BB 8
#define NN 1024
#define KNB 8
#define CC 128
#define BN (BB * NN)            // 8192 nodes
#define SROW ((size_t)BN * 128) // elements in one [BN,128] plane

typedef _Float16 v4h  __attribute__((ext_vector_type(4)));
typedef _Float16 v8h  __attribute__((ext_vector_type(8)));
typedef _Float16 v16h __attribute__((ext_vector_type(16)));
typedef float    v8f  __attribute__((ext_vector_type(8)));

// ---------------------------------------------------------------------------
// WMMA fragment loads from LDS (layouts per CDNA5 ISA 7.12.2, wave32)
// A: 16x32 f16. lane m=lane&15, half=lane>>4. K groups [8h..8h+7], [16+8h..+7]
// B: 32x16 f16 (stored as W^T[n][k] so per-lane K run is contiguous)
// ---------------------------------------------------------------------------
__device__ inline v16h load_a_frag(const _Float16* sA, int mBase, int k0, int lane) {
    int m = mBase + (lane & 15);
    int hf = lane >> 4;
    const _Float16* p = sA + m * 128 + k0 + 8 * hf;
    union { v16h v; v8h h[2]; } u;
    u.h[0] = *(const v8h*)(p);
    u.h[1] = *(const v8h*)(p + 16);
    return u.v;
}

__device__ inline v16h load_b_frag(const _Float16* sW, int n0, int k0, int lane) {
    int n = n0 + (lane & 15);
    int kh = lane >> 4;
    const _Float16* p = sW + n * 128 + k0 + 8 * kh;
    union { v16h v; v8h h[2]; } u;
    u.h[0] = *(const v8h*)(p);
    u.h[1] = *(const v8h*)(p + 16);
    return u.v;
}

// ---------------------------------------------------------------------------
// Generic 128x128x128 GEMM tile: one workgroup (8 waves) computes a
// [128 rows x 128 cols] block. Whole K=128 staged in LDS once.
// Staging is fully vectorized: 16B global loads -> 16B LDS stores
// (f32 sources get packed f32->f16 conversion in registers).
// SRC_F16: source activations are f16 (else f32, converted while staging)
// ACCUM:   dst is f32 and we do dst += acc + bias[col]; else dst is f16 store
// ---------------------------------------------------------------------------
template <bool SRC_F16, bool ACCUM>
__device__ inline void gemm128_tile(const void* src, const _Float16* wT,
                                    void* dst, const float* bias, int mBase) {
    __shared__ __align__(16) _Float16 sA[128 * 128];
    __shared__ __align__(16) _Float16 sW[128 * 128];
    int tid = threadIdx.x;

    // Weights: 32KB as 2048 x 16B chunks (8 per thread)
    {
        const uint4* wsrc = (const uint4*)wT;
        uint4* w4 = (uint4*)sW;
#pragma unroll
        for (int it = 0; it < 8; ++it) w4[tid + it * 256] = wsrc[tid + it * 256];
    }
    // Activations
    if (SRC_F16) {
        const uint4* asrc = (const uint4*)((const _Float16*)src + (size_t)mBase * 128);
        uint4* a4 = (uint4*)sA;
#pragma unroll
        for (int it = 0; it < 8; ++it) a4[tid + it * 256] = asrc[tid + it * 256];
    } else {
        const float4* asrc = (const float4*)((const float*)src + (size_t)mBase * 128);
        v4h* a4 = (v4h*)sA;
#pragma unroll
        for (int it = 0; it < 16; ++it) {
            float4 f = asrc[tid + it * 256];
            v4h h;
            h[0] = (_Float16)f.x; h[1] = (_Float16)f.y;
            h[2] = (_Float16)f.z; h[3] = (_Float16)f.w;
            a4[tid + it * 256] = h;
        }
    }
    __syncthreads();

    int wave = tid >> 5, lane = tid & 31;
    v8f zero = {0.f, 0.f, 0.f, 0.f, 0.f, 0.f, 0.f, 0.f};
    v8f acc[8];
#pragma unroll
    for (int nt = 0; nt < 8; ++nt) acc[nt] = zero;

#pragma unroll
    for (int kk = 0; kk < 4; ++kk) {
        v16h a = load_a_frag(sA, wave * 16, kk * 32, lane);
#pragma unroll
        for (int nt = 0; nt < 8; ++nt) {
            v16h b = load_b_frag(sW, nt * 16, kk * 32, lane);
            acc[nt] = __builtin_amdgcn_wmma_f32_16x16x32_f16(
                false, a, false, b, (short)0, acc[nt], false, false);
        }
    }

    // C/D layout: lane n = lane&15; rows (lane>>4)*8 + i
    int n = lane & 15, mt = (lane >> 4) * 8;
#pragma unroll
    for (int nt = 0; nt < 8; ++nt) {
        int col = nt * 16 + n;
#pragma unroll
        for (int i = 0; i < 8; ++i) {
            int row = mBase + wave * 16 + mt + i;
            if (ACCUM) {
                float* dF = (float*)dst;
                dF[(size_t)row * 128 + col] += acc[nt][i] + bias[col];
            } else {
                _Float16* dH = (_Float16*)dst;
                dH[(size_t)row * 128 + col] = (_Float16)acc[nt][i];
            }
        }
    }
}

// ---------------------------------------------------------------------------
// Per-layer projection GEMMs: 26 jobs of [BN,128]@[128,128] -> f16 proj planes
// p=0: q0=f0@W0 | p=1..3: q1_v=f1_v@W1 | p=4..7: f0@{W2,W4,W7,W9}
// p=8..25: f1_v @ {W3,W5,W6,W8,W10,W11} (3 comps each)
// ---------------------------------------------------------------------------
__global__ __launch_bounds__(256) void proj_gemm_kernel(
    const float* f0, const float* f1, const _Float16* WqkvT,
    _Float16* proj, int layer) {
    static const int w_of_p[26] = {0, 1, 1, 1, 2, 4, 7, 9,
                                   3, 3, 3, 5, 5, 5, 6, 6, 6,
                                   8, 8, 8, 10, 10, 10, 11, 11, 11};
    static const int s_of_p[26] = {0, 1, 2, 3, 0, 0, 0, 0,
                                   1, 2, 3, 1, 2, 3, 1, 2, 3,
                                   1, 2, 3, 1, 2, 3, 1, 2, 3};
    int p = blockIdx.y;
    int w = w_of_p[p], s = s_of_p[p];
    const float* src = (s == 0) ? f0 : (f1 + (size_t)(s - 1) * SROW);
    const _Float16* wT = WqkvT + ((size_t)layer * 12 + w) * 16384;
    _Float16* dst = proj + (size_t)p * SROW;
    gemm128_tile<false, false>(src, wT, dst, nullptr, blockIdx.x * 128);
}

// Output projections: f0 += o0@Wo0 + bo0 ; f1_v += o1_v@Wo1 + bo1
__global__ __launch_bounds__(256) void outproj_gemm_kernel(
    const _Float16* o0b, const _Float16* o1b, const _Float16* WoT,
    const float* bo, float* f0, float* f1, int layer) {
    int jy = blockIdx.y;
    const _Float16* src;
    const _Float16* wT;
    float* dst;
    const float* bias;
    if (jy == 0) {
        src = o0b;
        wT = WoT + (size_t)(layer * 2) * 16384;
        dst = f0;
        bias = bo + (size_t)(layer * 2) * 128;
    } else {
        src = o1b + (size_t)(jy - 1) * SROW;
        wT = WoT + (size_t)(layer * 2 + 1) * 16384;
        dst = f1 + (size_t)(jy - 1) * SROW;
        bias = bo + (size_t)(layer * 2 + 1) * 128;
    }
    gemm128_tile<true, true>(src, wT, dst, bias, blockIdx.x * 128);
}

// ---------------------------------------------------------------------------
// Weight prep: transpose + f16-convert Wqkv (48 mats) and Wo (8 mats)
// dstT[n][k] = src[k][n]
// ---------------------------------------------------------------------------
__global__ __launch_bounds__(256) void prep_weights_kernel(
    const float* Wqkv, const float* Wo, _Float16* WqkvT, _Float16* WoT) {
    int mat = blockIdx.x;
    const float* src;
    _Float16* dst;
    if (mat < 48) { src = Wqkv + (size_t)mat * 16384; dst = WqkvT + (size_t)mat * 16384; }
    else          { src = Wo + (size_t)(mat - 48) * 16384; dst = WoT + (size_t)(mat - 48) * 16384; }
    for (int t = threadIdx.x; t < 16384; t += 256) {
        int r = t >> 7, c = t & 127;
        dst[c * 128 + r] = (_Float16)src[t];
    }
}

// ---------------------------------------------------------------------------
// Time embedding + f0/f1 init. One block (128 thr) per batch.
// ---------------------------------------------------------------------------
__global__ __launch_bounds__(128) void init_embed_kernel(
    const float* x, const float* t, const float* Wt1, const float* bt1,
    const float* Wt2, const float* bt2, const float* Win0, const float* bin0,
    const float* Win1, float* f0, float* f1) {
    __shared__ float s0[128], s1[128], s2[64], sBase[128], sWx0[128], sWx1[128], sWx2[128], sW1c[128];
    int b = blockIdx.x, c = threadIdx.x;
    float tb = t[b];
    if (c < 64) {
        float fr = __expf(-logf(10000.f) * (float)c / 64.f);
        float e = tb * fr;
        s0[c] = sinf(e);
        s0[c + 64] = cosf(e);
    }
    __syncthreads();
    {
        float a = bt1[c];
        for (int i = 0; i < 128; ++i) a += s0[i] * Wt1[i * 128 + c];
        s1[c] = a / (1.f + __expf(-a)); // silu
    }
    __syncthreads();
    if (c < 64) {
        float a = bt2[c];
        for (int i = 0; i < 128; ++i) a += s1[i] * Wt2[i * 64 + c];
        s2[c] = a;
    }
    __syncthreads();
    {
        float a = bin0[c];
        for (int i = 0; i < 64; ++i) a += s2[i] * Win0[i * 128 + c];
        sBase[c] = a;
        sWx0[c] = Win0[64 * 128 + c];
        sWx1[c] = Win0[65 * 128 + c];
        sWx2[c] = Win0[66 * 128 + c];
        sW1c[c] = Win1[c];
    }
    __syncthreads();
    size_t nodeBase = (size_t)b * NN;
    for (int n = 0; n < NN; ++n) {
        size_t node = nodeBase + n;
        float x0 = x[node * 3 + 0], x1 = x[node * 3 + 1], x2 = x[node * 3 + 2];
        size_t o = node * 128 + c;
        f0[o] = sBase[c] + x0 * sWx0[c] + x1 * sWx1[c] + x2 * sWx2[c];
        f1[o] = x0 * sW1c[c];
        f1[SROW + o] = x1 * sW1c[c];
        f1[2 * SROW + o] = x2 * sW1c[c];
    }
}

// ---------------------------------------------------------------------------
// KNN top-8 per node (exclude self), plus r / rhat. Block = 256 queries of
// one batch; all batch positions staged in LDS.
// ---------------------------------------------------------------------------
__global__ __launch_bounds__(256) void knn_kernel(const float* x, int* idxb,
                                                  float* rb, float* rhb) {
    __shared__ float sX[NN * 3];
    int b = blockIdx.y, tid = threadIdx.x;
    for (int i = tid; i < NN * 3; i += 256) sX[i] = x[(size_t)b * NN * 3 + i];
    __syncthreads();
    int n = blockIdx.x * 256 + tid;
    float qx = sX[n * 3], qy = sX[n * 3 + 1], qz = sX[n * 3 + 2];
    float bd[8];
    int bi[8];
#pragma unroll
    for (int k = 0; k < 8; ++k) { bd[k] = 3.0e38f; bi[k] = 0; }
    for (int m = 0; m < NN; ++m) {
        float dx = sX[m * 3] - qx, dy = sX[m * 3 + 1] - qy, dz = sX[m * 3 + 2] - qz;
        float d2 = dx * dx + dy * dy + dz * dz;
        if (m == n) continue; // reference adds 1e9 on diagonal
        if (d2 < bd[7]) {
            bd[7] = d2; bi[7] = m;
#pragma unroll
            for (int q = 7; q > 0; --q) {
                if (bd[q] < bd[q - 1]) {
                    float td = bd[q]; bd[q] = bd[q - 1]; bd[q - 1] = td;
                    int ti = bi[q]; bi[q] = bi[q - 1]; bi[q - 1] = ti;
                }
            }
        }
    }
    size_t base = ((size_t)b * NN + n) * 8;
#pragma unroll
    for (int k = 0; k < 8; ++k) {
        int j = bi[k];
        idxb[base + k] = j;
        float rx = sX[j * 3] - qx, ry = sX[j * 3 + 1] - qy, rz = sX[j * 3 + 2] - qz;
        float rr = sqrtf(rx * rx + ry * ry + rz * rz + 1e-8f);
        rb[base + k] = rr;
        float iv = 1.f / rr;
        rhb[(base + k) * 3 + 0] = rx * iv;
        rhb[(base + k) * 3 + 1] = ry * iv;
        rhb[(base + k) * 3 + 2] = rz * iv;
    }
}

// ---------------------------------------------------------------------------
// Attention combine: one wave per node. Lane covers hd = 4*lane..4*lane+3,
// head h = lane>>2; 4-lane shuffle reduction for per-head dots.
// Pass 1 computes logits and prefetches the pass-2 (value) rows.
// ---------------------------------------------------------------------------
__device__ inline void load4(const _Float16* p, float* o) {
    v4h v = *(const v4h*)p;
    o[0] = (float)v[0]; o[1] = (float)v[1]; o[2] = (float)v[2]; o[3] = (float)v[3];
}

__global__ __launch_bounds__(256) void attention_kernel(
    int layer, const _Float16* proj, const int* idxb, const float* rb,
    const float* rhb, const float* Wr1, const float* br1, const float* Wr2,
    const float* br2, _Float16* o0b, _Float16* o1b) {
    __shared__ float sWr1[32], sbr1[32], sWr2[32 * 80], sbr2[80];
    __shared__ float sRn[8][8][80];
    int tid = threadIdx.x;
    if (tid < 32) { sWr1[tid] = Wr1[layer * 32 + tid]; sbr1[tid] = br1[layer * 32 + tid]; }
    for (int i = tid; i < 2560; i += 256) sWr2[i] = Wr2[layer * 2560 + i];
    if (tid < 80) sbr2[tid] = br2[layer * 80 + tid];
    __syncthreads();

    int wave = tid >> 5, lane = tid & 31;
    int node = blockIdx.x * 8 + wave;

    // Rn[k][h*10+p] = silu(r*Wr1+br1) @ Wr2 + br2
    for (int tt = lane; tt < 640; tt += 32) {
        int k = tt / 80, hp = tt % 80;
        float rk = rb[node * 8 + k];
        float a = sbr2[hp];
        for (int i = 0; i < 32; ++i) {
            float hh = rk * sWr1[i] + sbr1[i];
            a += (hh / (1.f + __expf(-hh))) * sWr2[i * 80 + hp];
        }
        sRn[wave][k][hp] = a;
    }
    __syncthreads();

    int hd0 = lane * 4;
    int h = lane >> 2;
    float q0r[4], q1r[3][4];
    load4(proj + (size_t)node * 128 + hd0, q0r);
#pragma unroll
    for (int v = 0; v < 3; ++v)
        load4(proj + (size_t)(1 + v) * SROW + (size_t)node * 128 + hd0, q1r[v]);

    float lk[8];
#pragma unroll
    for (int k = 0; k < 8; ++k) {
        int j = idxb[node * 8 + k];
        float r0 = rhb[(node * 8 + k) * 3 + 0];
        float r1 = rhb[(node * 8 + k) * 3 + 1];
        float r2 = rhb[(node * 8 + k) * 3 + 2];
        size_t jo = (size_t)j * 128 + hd0;

        // Prefetch pass-2 (value) rows for this neighbor while we do math
        // (lowers to global_prefetch_b8 on gfx1250).
        __builtin_prefetch(proj + 6 * SROW + jo, 0, 1);
        __builtin_prefetch(proj + 7 * SROW + jo, 0, 1);
#pragma unroll
        for (int v = 0; v < 3; ++v) {
            __builtin_prefetch(proj + (size_t)(17 + v) * SROW + jo, 0, 1);
            __builtin_prefetch(proj + (size_t)(20 + v) * SROW + jo, 0, 1);
            __builtin_prefetch(proj + (size_t)(23 + v) * SROW + jo, 0, 1);
        }

        const float* Rn = &sRn[wave][k][h * 10];
        float R0 = Rn[0], R1 = Rn[1], R2 = Rn[2], R3 = Rn[3], R4 = Rn[4];
        float g2e[4], g4e[4], h3e[3][4], h5e[3][4], h6e[3][4];
        load4(proj + 4 * SROW + jo, g2e);
        load4(proj + 5 * SROW + jo, g4e);
#pragma unroll
        for (int v = 0; v < 3; ++v) {
            load4(proj + (size_t)(8 + v) * SROW + jo, h3e[v]);
            load4(proj + (size_t)(11 + v) * SROW + jo, h5e[v]);
            load4(proj + (size_t)(14 + v) * SROW + jo, h6e[v]);
        }
        float rhv[3] = {r0, r1, r2};
        float part = 0.f;
#pragma unroll
        for (int jj = 0; jj < 4; ++jj) {
            float d3 = r0 * h3e[0][jj] + r1 * h3e[1][jj] + r2 * h3e[2][jj];
            part += q0r[jj] * (g2e[jj] * R0 + d3 * R1);
            float d6 = r0 * h6e[0][jj] + r1 * h6e[1][jj] + r2 * h6e[2][jj];
#pragma unroll
            for (int v = 0; v < 3; ++v) {
                float k1 = g4e[jj] * rhv[v] * R2 + h5e[v][jj] * R3 +
                           (3.f * rhv[v] * d6 - h6e[v][jj]) * R4;
                part += q1r[v][jj] * k1;
            }
        }
        part += __shfl_xor(part, 1, 32);
        part += __shfl_xor(part, 2, 32);
        lk[k] = part * 0.17677669529663687f; // 1/sqrt(2*Dh)
    }

    float mx = lk[0];
#pragma unroll
    for (int k = 1; k < 8; ++k) mx = fmaxf(mx, lk[k]);
    float aw[8], ssum = 0.f;
#pragma unroll
    for (int k = 0; k < 8; ++k) { aw[k] = __expf(lk[k] - mx); ssum += aw[k]; }
    float inv = 1.f / ssum;
#pragma unroll
    for (int k = 0; k < 8; ++k) aw[k] *= inv;

    float o0a[4] = {0.f, 0.f, 0.f, 0.f};
    float o1a[3][4] = {};
#pragma unroll
    for (int k = 0; k < 8; ++k) {
        int j = idxb[node * 8 + k];
        float r0 = rhb[(node * 8 + k) * 3 + 0];
        float r1 = rhb[(node * 8 + k) * 3 + 1];
        float r2 = rhb[(node * 8 + k) * 3 + 2];
        const float* Rn = &sRn[wave][k][h * 10];
        float R5 = Rn[5], R6 = Rn[6], R7 = Rn[7], R8 = Rn[8], R9 = Rn[9];
        size_t jo = (size_t)j * 128 + hd0;
        float g7e[4], g9e[4], h8e[3][4], h10e[3][4], h11e[3][4];
        load4(proj + 6 * SROW + jo, g7e);
        load4(proj + 7 * SROW + jo, g9e);
#pragma unroll
        for (int v = 0; v < 3; ++v) {
            load4(proj + (size_t)(17 + v) * SROW + jo, h8e[v]);
            load4(proj + (size_t)(20 + v) * SROW + jo, h10e[v]);
            load4(proj + (size_t)(23 + v) * SROW + jo, h11e[v]);
        }
        float rhv[3] = {r0, r1, r2};
        float a = aw[k];
#pragma unroll
        for (int jj = 0; jj < 4; ++jj) {
            float d8 = r0 * h8e[0][jj] + r1 * h8e[1][jj] + r2 * h8e[2][jj];
            o0a[jj] += a * (g7e[jj] * R5 + d8 * R6);
            float d11 = r0 * h11e[0][jj] + r1 * h11e[1][jj] + r2 * h11e[2][jj];
#pragma unroll
            for (int v = 0; v < 3; ++v) {
                float v1 = g9e[jj] * rhv[v] * R7 + h10e[v][jj] * R8 +
                           (3.f * rhv[v] * d11 - h11e[v][jj]) * R9;
                o1a[v][jj] += a * v1;
            }
        }
    }

    size_t oo = (size_t)node * 128 + hd0;
#pragma unroll
    for (int jj = 0; jj < 4; ++jj) o0b[oo + jj] = (_Float16)o0a[jj];
#pragma unroll
    for (int v = 0; v < 3; ++v)
#pragma unroll
        for (int jj = 0; jj < 4; ++jj)
            o1b[(size_t)v * SROW + oo + jj] = (_Float16)o1a[v][jj];
}

// ---------------------------------------------------------------------------
// Final: out[b,n,v] = sum_c f1[v][node][c]*Wout[v][c] + bout[v]
// ---------------------------------------------------------------------------
__global__ __launch_bounds__(256) void final_out_kernel(
    const float* f1, const float* Wout, const float* bout, float* out) {
    int node = blockIdx.x * 256 + threadIdx.x;
#pragma unroll
    for (int v = 0; v < 3; ++v) {
        float a = bout[v];
        const float* row = f1 + (size_t)v * SROW + (size_t)node * 128;
        for (int c = 0; c < 128; ++c) a += row[c] * Wout[v * 128 + c];
        out[(size_t)node * 3 + v] = a;
    }
}

// ---------------------------------------------------------------------------
extern "C" void kernel_launch(void* const* d_in, const int* in_sizes, int n_in,
                              void* d_out, int out_size, void* d_ws, size_t ws_size,
                              hipStream_t stream) {
    const float* x    = (const float*)d_in[0];
    const float* t    = (const float*)d_in[2];
    const float* Wt1  = (const float*)d_in[3];
    const float* bt1  = (const float*)d_in[4];
    const float* Wt2  = (const float*)d_in[5];
    const float* bt2  = (const float*)d_in[6];
    const float* Win0 = (const float*)d_in[7];
    const float* bin0 = (const float*)d_in[8];
    const float* Win1 = (const float*)d_in[9];
    const float* Wqkv = (const float*)d_in[10];
    const float* Wr1  = (const float*)d_in[11];
    const float* br1  = (const float*)d_in[12];
    const float* Wr2  = (const float*)d_in[13];
    const float* br2  = (const float*)d_in[14];
    const float* Wo   = (const float*)d_in[15];
    const float* bo   = (const float*)d_in[16];
    const float* Wout = (const float*)d_in[17];
    const float* bout = (const float*)d_in[18];

    char* ws = (char*)d_ws;
    size_t off = 0;
    auto wsAlloc = [&](size_t bytes) -> void* {
        void* p = (void*)(ws + off);
        off += (bytes + 255) & ~(size_t)255;
        return p;
    };
    float*     f0    = (float*)wsAlloc((size_t)BN * 128 * 4);
    float*     f1    = (float*)wsAlloc((size_t)3 * BN * 128 * 4);
    int*       idxb  = (int*)wsAlloc((size_t)BN * 8 * 4);
    float*     rb    = (float*)wsAlloc((size_t)BN * 8 * 4);
    float*     rhb   = (float*)wsAlloc((size_t)BN * 8 * 3 * 4);
    _Float16*  WqkvT = (_Float16*)wsAlloc((size_t)48 * 16384 * 2);
    _Float16*  WoT   = (_Float16*)wsAlloc((size_t)8 * 16384 * 2);
    _Float16*  proj  = (_Float16*)wsAlloc((size_t)26 * BN * 128 * 2);
    _Float16*  o0b   = (_Float16*)wsAlloc((size_t)BN * 128 * 2);
    _Float16*  o1b   = (_Float16*)wsAlloc((size_t)3 * BN * 128 * 2);
    (void)ws_size; (void)in_sizes; (void)n_in; (void)out_size;

    prep_weights_kernel<<<56, 256, 0, stream>>>(Wqkv, Wo, WqkvT, WoT);
    init_embed_kernel<<<BB, 128, 0, stream>>>(x, t, Wt1, bt1, Wt2, bt2,
                                              Win0, bin0, Win1, f0, f1);
    knn_kernel<<<dim3(NN / 256, BB), 256, 0, stream>>>(x, idxb, rb, rhb);

    for (int l = 0; l < 4; ++l) {
        proj_gemm_kernel<<<dim3(BN / 128, 26), 256, 0, stream>>>(f0, f1, WqkvT, proj, l);
        attention_kernel<<<BN / 8, 256, 0, stream>>>(l, proj, idxb, rb, rhb,
                                                     Wr1, br1, Wr2, br2, o0b, o1b);
        outproj_gemm_kernel<<<dim3(BN / 128, 4), 256, 0, stream>>>(o0b, o1b, WoT, bo,
                                                                   f0, f1, l);
    }
    final_out_kernel<<<BN / 256, 256, 0, stream>>>(f1, Wout, bout, (float*)d_out);
}
